// SoftSOMLayer_52329881534816
// MI455X (gfx1250) — matrix operand order repaired and checked by reading, not compiled, not run on hardware
//
#include <hip/hip_runtime.h>

typedef __attribute__((ext_vector_type(16))) _Float16 v16h;
typedef __attribute__((ext_vector_type(8)))  _Float16 v8h;
typedef __attribute__((ext_vector_type(4)))  _Float16 v4h;
typedef __attribute__((ext_vector_type(8)))  float    v8f;

#define D_DIM 128
#define M_DIM 256
#define NEG_INV_T (-2.5f)   // -1/0.4

#define CAT16(lo, hi) __builtin_shufflevector(lo, hi, 0,1,2,3,4,5,6,7,8,9,10,11,12,13,14,15)

// ---------------------------------------------------------------------------
// Prep: psq[m] = ||P[m]||^2 (fp32), P16 = f16 row-major P, Pt16 = f16 P^T.
// ---------------------------------------------------------------------------
__global__ void prep_kernel(const float* __restrict__ P,
                            float* __restrict__ psq,
                            _Float16* __restrict__ P16,
                            _Float16* __restrict__ Pt16) {
    const int m = threadIdx.x;                 // blockDim.x == 256
    float s = 0.f;
#pragma unroll 4
    for (int k = 0; k < D_DIM; ++k) {
        const float v = P[m * D_DIM + k];
        s += v * v;
        const _Float16 hv = (_Float16)v;
        P16[m * D_DIM + k]  = hv;              // [256][128] row-major
        Pt16[k * M_DIM + m] = hv;              // [128][256] transposed
    }
    psq[m] = s;
}

// ---------------------------------------------------------------------------
// Fused: cross-GEMM (f16 WMMA, fp32 acc) -> dist -> fp32 softmax -> blend-GEMM.
// block = 64 threads = 2 wave32; each wave owns a 16-token tile.
// ---------------------------------------------------------------------------
__launch_bounds__(64)
__global__ void som_kernel(const float* __restrict__ X,
                           const _Float16* __restrict__ P16,
                           const _Float16* __restrict__ Pt16,
                           const float* __restrict__ psq,
                           float* __restrict__ outB,    // [N,128]
                           float* __restrict__ outW) {  // [N,256]
    __shared__ float    scores[2][16][264];   // fp32 scores, padded rows (33 KB)
    __shared__ _Float16 wh[2][16][272];       // f16 weights for GEMM2-A (17 KB)

    const int wave   = threadIdx.x >> 5;
    const int lane   = threadIdx.x & 31;
    const int lane16 = lane & 15;
    const int half   = lane >> 4;
    const long tok0  = (long)blockIdx.x * 32 + wave * 16;

    // ---- X tile: build f16 A-fragments (16x32 layout) + fp32 ||x||^2 ----
    // lane-half h owns K = kb*32 + {8h..8h+7} (elems 0-7) and kb*32+16+{8h..8h+7} (8-15)
    v16h a[4];
    float ps = 0.f;
    const float* xrow = X + (tok0 + lane16) * D_DIM;
#pragma unroll
    for (int kb = 0; kb < 4; ++kb) {
        const int c1 = kb * 32 + 8 * half;
        float4 f0 = *(const float4*)(xrow + c1);
        float4 f1 = *(const float4*)(xrow + c1 + 4);
        float4 f2 = *(const float4*)(xrow + c1 + 16);
        float4 f3 = *(const float4*)(xrow + c1 + 20);
        const float t[16] = {f0.x,f0.y,f0.z,f0.w, f1.x,f1.y,f1.z,f1.w,
                             f2.x,f2.y,f2.z,f2.w, f3.x,f3.y,f3.z,f3.w};
#pragma unroll
        for (int e = 0; e < 16; ++e) { ps += t[e] * t[e]; a[kb][e] = (_Float16)t[e]; }
    }
    ps += __shfl_xor(ps, 16, 32);             // full ||x||^2 for row lane16 (wave32)
    float xr[8];
#pragma unroll
    for (int v = 0; v < 8; ++v) xr[v] = __shfl(ps, v + half * 8, 32);

    // ---- GEMM1: cross = X @ P^T via wmma_f32_16x16x32_f16 ----------------
#pragma unroll 1
    for (int t = 0; t < 16; ++t) {
        v8f acc = {};
        // B 32x16: lane-half h, col lane16 owns K = kb*32+{8h..8h+7} and kb*32+16+{8h..8h+7}
        const _Float16* prow = P16 + (t * 16 + lane16) * D_DIM + 8 * half;
#pragma unroll
        for (int kb = 0; kb < 4; ++kb) {
            v8h blo = *(const v8h*)(prow + kb * 32);
            v8h bhi = *(const v8h*)(prow + kb * 32 + 16);
            v16h b  = CAT16(blo, bhi);
            acc = __builtin_amdgcn_wmma_f32_16x16x32_f16(
                false, a[kb], false, b, (short)0, acc, false, false);
        }
        const float pq = psq[t * 16 + lane16];
#pragma unroll
        for (int v = 0; v < 8; ++v) {          // C layout: VGPR v -> row v+8*half
            float d2 = fmaxf(xr[v] + pq - 2.0f * acc[v], 0.0f);
            scores[wave][v + half * 8][t * 16 + lane16] = __builtin_sqrtf(d2) * NEG_INV_T;
        }
    }
    __syncthreads();

    // ---- Row softmax (fp32, wave32 shuffle reductions, vector LDS/VMEM) --
#pragma unroll 1
    for (int r = 0; r < 16; ++r) {
        float4 u0 = *(const float4*)&scores[wave][r][lane * 4];
        float4 u1 = *(const float4*)&scores[wave][r][128 + lane * 4];
        float m = fmaxf(fmaxf(fmaxf(u0.x, u0.y), fmaxf(u0.z, u0.w)),
                        fmaxf(fmaxf(u1.x, u1.y), fmaxf(u1.z, u1.w)));
#pragma unroll
        for (int off = 16; off > 0; off >>= 1) m = fmaxf(m, __shfl_xor(m, off, 32));
        float e0 = __expf(u0.x - m), e1 = __expf(u0.y - m);
        float e2 = __expf(u0.z - m), e3 = __expf(u0.w - m);
        float e4 = __expf(u1.x - m), e5 = __expf(u1.y - m);
        float e6 = __expf(u1.z - m), e7 = __expf(u1.w - m);
        float s = ((e0 + e1) + (e2 + e3)) + ((e4 + e5) + (e6 + e7));
#pragma unroll
        for (int off = 16; off > 0; off >>= 1) s += __shfl_xor(s, off, 32);
        const float inv = 1.0f / s;
        float4 w0 = {e0 * inv, e1 * inv, e2 * inv, e3 * inv};
        float4 w1 = {e4 * inv, e5 * inv, e6 * inv, e7 * inv};
        // main HBM stream: coalesced b128 weight stores
        *(float4*)(outW + (tok0 + r) * M_DIM + lane * 4)       = w0;
        *(float4*)(outW + (tok0 + r) * M_DIM + 128 + lane * 4) = w1;
        v4h h0 = {(_Float16)w0.x, (_Float16)w0.y, (_Float16)w0.z, (_Float16)w0.w};
        v4h h1 = {(_Float16)w1.x, (_Float16)w1.y, (_Float16)w1.z, (_Float16)w1.w};
        *(v4h*)&wh[wave][r][lane * 4]       = h0;
        *(v4h*)&wh[wave][r][128 + lane * 4] = h1;
    }
    __syncthreads();

    // ---- GEMM2: blended = W(16x256) @ P(256x128) -------------------------
    // A-fragments (f16 weights) hoisted: reused across all 8 N-tiles.
    v16h aw[8];
    const _Float16* wrow = &wh[wave][lane16][8 * half];
#pragma unroll
    for (int kb = 0; kb < 8; ++kb) {
        v8h lo = *(const v8h*)(wrow + kb * 32);        // ds_load_b128
        v8h hi = *(const v8h*)(wrow + kb * 32 + 16);
        aw[kb] = CAT16(lo, hi);
    }
#pragma unroll 1
    for (int n0 = 0; n0 < 8; ++n0) {
        v8f acc = {};
        const _Float16* pcol = Pt16 + (n0 * 16 + lane16) * M_DIM + 8 * half;
#pragma unroll
        for (int kb = 0; kb < 8; ++kb) {
            v8h lo = *(const v8h*)(pcol + kb * 32);
            v8h hi = *(const v8h*)(pcol + kb * 32 + 16);
            v16h b = CAT16(lo, hi);
            acc = __builtin_amdgcn_wmma_f32_16x16x32_f16(
                false, aw[kb], false, b, (short)0, acc, false, false);
        }
#pragma unroll
        for (int v = 0; v < 8; ++v)
            outB[(tok0 + v + half * 8) * D_DIM + n0 * 16 + lane16] = acc[v];
    }
}

// ---------------------------------------------------------------------------
extern "C" void kernel_launch(void* const* d_in, const int* in_sizes, int n_in,
                              void* d_out, int out_size, void* d_ws, size_t ws_size,
                              hipStream_t stream) {
    const float* x = (const float*)d_in[0];        // [N,128] fp32
    const float* P = (const float*)d_in[1];        // [256,128] fp32
    const int N = in_sizes[0] / D_DIM;

    // workspace layout: psq (1 KB) | P16 (64 KB) | Pt16 (64 KB)
    float*    psq  = (float*)d_ws;
    _Float16* P16  = (_Float16*)((char*)d_ws + 1024);
    _Float16* Pt16 = (_Float16*)((char*)d_ws + 1024 + M_DIM * D_DIM * 2);

    float* outB = (float*)d_out;                   // [N,128]
    float* outW = outB + (size_t)N * D_DIM;        // [N,256]

    prep_kernel<<<1, 256, 0, stream>>>(P, psq, P16, Pt16);
    som_kernel<<<N / 32, 64, 0, stream>>>(x, P16, Pt16, psq, outB, outW);
}
